// GRU_50062138802351
// MI455X (gfx1250) — matrix operand order repaired and checked by reading, not compiled, not run on hardware
//
#include <hip/hip_runtime.h>

// ---------------------------------------------------------------------------
// GRU cell, B=8192 H=2048, fp32 ref -> bf16 WMMA (f32 accum) on gfx1250.
// Phase 0: convert x,ht -> bf16; convert+transpose 6 weights -> bf16 [n][k].
// Phase 1: gate GEMMs (z, r*ht, xh), async-to-LDS double buffering.
// Phase 2: (r*ht)@Uh + epilogue -> out.
// Block tile 128x256, 8 waves, 64x64 wave tile, BK=64 stages.
// ---------------------------------------------------------------------------

typedef __attribute__((ext_vector_type(16))) __bf16 v16bf;
typedef __attribute__((ext_vector_type(8)))  float  v8f;
typedef int vi4 __attribute__((vector_size(16)));

#define BDIM 8192
#define HDIM 2048
#define BM 128
#define BN 256
#define BK 64
#define LDSS 72          // LDS row stride in shorts (BK + 8 pad; 144B, 16B-aligned)
#define NTHREADS 256

#if defined(__HIP_DEVICE_COMPILE__) && __has_builtin(__builtin_amdgcn_global_load_async_to_lds_b128)
#define HAS_ASYNC 1
#else
#define HAS_ASYNC 0
#endif

union Frag16 { unsigned int u[8]; v16bf v; };

__device__ __forceinline__ unsigned short f2bf(float f) {
    unsigned int u = __float_as_uint(f);
    unsigned int r = u + 0x7FFFu + ((u >> 16) & 1u);   // round-to-nearest-even
    return (unsigned short)(r >> 16);
}
__device__ __forceinline__ unsigned int pack2(float a, float b) {
    return (unsigned int)f2bf(a) | ((unsigned int)f2bf(b) << 16);
}

__device__ __forceinline__ void async_wait0() {
#if defined(__HIP_DEVICE_COMPILE__)
#if __has_builtin(__builtin_amdgcn_s_wait_asynccnt)
    __builtin_amdgcn_s_wait_asynccnt(0);
#else
    asm volatile("s_wait_asynccnt 0x0" ::: "memory");
#endif
#endif
}

// ---------------------------------------------------------------------------
// Tile movers: ROWS x 64 bf16 (128B/row) = ROWS*8 16B chunks over 256 threads.
// Global row stride = HDIM shorts, LDS stride = LDSS.
// ---------------------------------------------------------------------------
#if HAS_ASYNC
template <int ROWS>
__device__ __forceinline__ void async_tile_copy(const unsigned short* __restrict__ g,
                                                unsigned short* __restrict__ s, int tid) {
#pragma unroll
    for (int i = 0; i < (ROWS * 8) / NTHREADS; ++i) {
        int idx = tid + i * NTHREADS;
        int row = idx >> 3;
        int c   = (idx & 7) * 8;                // short offset, 16B chunks
        const unsigned short* gp = g + (size_t)row * HDIM + c;
        unsigned short* lp = s + row * LDSS + c;
        __builtin_amdgcn_global_load_async_to_lds_b128(
            (__attribute__((address_space(1))) vi4*)gp,
            (__attribute__((address_space(3))) vi4*)lp, 0, 0);
    }
}
#else
template <int ROWS>
__device__ __forceinline__ void load_tile_regs(const unsigned short* __restrict__ g,
                                               int tid, uint4* r) {
#pragma unroll
    for (int i = 0; i < (ROWS * 8) / NTHREADS; ++i) {
        int idx = tid + i * NTHREADS;
        int row = idx >> 3;
        int c   = (idx & 7) * 8;
        r[i] = *(const uint4*)(g + (size_t)row * HDIM + c);
    }
}
template <int ROWS>
__device__ __forceinline__ void store_tile_regs(unsigned short* __restrict__ s,
                                                int tid, const uint4* r) {
#pragma unroll
    for (int i = 0; i < (ROWS * 8) / NTHREADS; ++i) {
        int idx = tid + i * NTHREADS;
        int row = idx >> 3;
        int c   = (idx & 7) * 8;
        *(uint4*)(s + row * LDSS + c) = r[i];
    }
}
#endif

// ---------------------------------------------------------------------------
// One BK=64 stage: two K=32 slabs, 16 WMMAs each, per wave (64x64 wave tile).
// ---------------------------------------------------------------------------
__device__ __forceinline__ void mma_stage(const unsigned short* __restrict__ sA,
                                          const unsigned short* __restrict__ sB,
                                          int wm, int wn, int lane, v8f acc[4][4]) {
    const int m0 = lane & 15;
    const int hi = lane >> 4;
#pragma unroll
    for (int s = 0; s < 2; ++s) {
        const int kbA = s * 32 + hi * 16;  // A: lanes 0-15 K=0-15; 16-31 K=16-31
        const int kbB = s * 32 + hi * 16;  // B: lanes 0-15 K=0-15; 16-31 K=16-31
        Frag16 bfr[4];
#pragma unroll
        for (int tj = 0; tj < 4; ++tj) {
            const unsigned short* bp = sB + (wn * 64 + tj * 16 + m0) * LDSS + kbB;
#pragma unroll
            for (int j = 0; j < 8; ++j)
                bfr[tj].u[j] = *(const unsigned int*)(bp + 2 * j);
        }
#pragma unroll
        for (int ti = 0; ti < 4; ++ti) {
            Frag16 afr;
            const unsigned short* ap = sA + (wm * 64 + ti * 16 + m0) * LDSS;
#pragma unroll
            for (int j = 0; j < 4; ++j)
                afr.u[j] = *(const unsigned int*)(ap + kbA + 2 * j);
#pragma unroll
            for (int j = 0; j < 4; ++j)
                afr.u[4 + j] = *(const unsigned int*)(ap + kbA + 8 + 2 * j);
#pragma unroll
            for (int tj = 0; tj < 4; ++tj)
                acc[ti][tj] = __builtin_amdgcn_wmma_f32_16x16x32_bf16(
                    false, afr.v, false, bfr[tj].v,
                    (short)0, acc[ti][tj], false, false);
        }
    }
}

// ---------------------------------------------------------------------------
// Full-K GEMM accumulate with double-buffered LDS. A: bf16 [M][K] stride HDIM.
// Bt: bf16 pre-transposed [N][K] stride HDIM.
// ---------------------------------------------------------------------------
__device__ __forceinline__ void gemm_accum(const unsigned short* __restrict__ A,
                                           const unsigned short* __restrict__ Bt,
                                           int br0, int bc0,
                                           unsigned short* __restrict__ sA,
                                           unsigned short* __restrict__ sB,
                                           v8f acc[4][4]) {
    const int tid  = threadIdx.x;
    const int lane = tid & 31;
    const int w    = tid >> 5;
    const int wm = w >> 2, wn = w & 3;
    const unsigned short* Ab = A + (size_t)br0 * HDIM;
    const unsigned short* Bb = Bt + (size_t)bc0 * HDIM;
    const int TILE_A = BM * LDSS;
    const int TILE_B = BN * LDSS;

#if HAS_ASYNC
    async_tile_copy<BM>(Ab, sA, tid);
    async_tile_copy<BN>(Bb, sB, tid);
    async_wait0();
    __syncthreads();
    int cur = 0;
    for (int kk = 0; kk < HDIM; kk += BK) {
        const int nxt = cur ^ 1;
        if (kk + BK < HDIM) {
            async_tile_copy<BM>(Ab + kk + BK, sA + nxt * TILE_A, tid);
            async_tile_copy<BN>(Bb + kk + BK, sB + nxt * TILE_B, tid);
        }
        mma_stage(sA + cur * TILE_A, sB + cur * TILE_B, wm, wn, lane, acc);
        async_wait0();
        __syncthreads();
        cur = nxt;
    }
#else
    uint4 pa[(BM * 8) / NTHREADS], pb[(BN * 8) / NTHREADS];
    load_tile_regs<BM>(Ab, tid, pa);
    load_tile_regs<BN>(Bb, tid, pb);
    store_tile_regs<BM>(sA, tid, pa);
    store_tile_regs<BN>(sB, tid, pb);
    __syncthreads();
    int cur = 0;
    for (int kk = 0; kk < HDIM; kk += BK) {
        const int nxt = cur ^ 1;
        const bool has_next = (kk + BK < HDIM);
        if (has_next) {
            load_tile_regs<BM>(Ab + kk + BK, tid, pa);
            load_tile_regs<BN>(Bb + kk + BK, tid, pb);
        }
        mma_stage(sA + cur * TILE_A, sB + cur * TILE_B, wm, wn, lane, acc);
        __syncthreads();
        if (has_next) {
            store_tile_regs<BM>(sA + nxt * TILE_A, tid, pa);
            store_tile_regs<BN>(sB + nxt * TILE_B, tid, pb);
        }
        __syncthreads();
        cur = nxt;
    }
#endif
}

// ---------------------------------------------------------------------------
// Phase-0 kernels: bf16 conversion / weight transpose.
// ---------------------------------------------------------------------------
__global__ __launch_bounds__(NTHREADS) void convert_bf16_kernel(
    const float* __restrict__ in, unsigned short* __restrict__ out, int n4) {
    int i = blockIdx.x * NTHREADS + threadIdx.x;
    if (i < n4) {
        float4 v = ((const float4*)in)[i];
        uint2 o;
        o.x = pack2(v.x, v.y);
        o.y = pack2(v.z, v.w);
        ((uint2*)out)[i] = o;
    }
}

#define TT 64
__global__ __launch_bounds__(NTHREADS) void transpose_bf16_kernel(
    const float* __restrict__ W, unsigned short* __restrict__ T) {
    __shared__ float tile[TT][TT + 1];
    const int k0 = blockIdx.y * TT;
    const int n0 = blockIdx.x * TT;
    const int tid = threadIdx.x;
#pragma unroll
    for (int i = 0; i < 4; ++i) {           // 64x64 f32 in, coalesced float4
        int idx = tid + i * NTHREADS;
        int r   = idx >> 4;
        int c4  = (idx & 15) * 4;
        float4 v = *(const float4*)(W + (size_t)(k0 + r) * HDIM + n0 + c4);
        tile[r][c4 + 0] = v.x; tile[r][c4 + 1] = v.y;
        tile[r][c4 + 2] = v.z; tile[r][c4 + 3] = v.w;
    }
    __syncthreads();
#pragma unroll
    for (int i = 0; i < 8; ++i) {           // 64x64 bf16 out, coalesced u32
        int idx = tid + i * NTHREADS;
        int n   = idx >> 5;
        int kp  = (idx & 31) * 2;
        unsigned int pv = pack2(tile[kp][n], tile[kp + 1][n]);
        *(unsigned int*)(T + (size_t)(n0 + n) * HDIM + k0 + kp) = pv;
    }
}

// ---------------------------------------------------------------------------
// Phase 1: per-gate GEMMs.
//   gate 0: ws_z   = sigmoid(x*Wz + ht*Uz + bz)              (f32)
//   gate 1: ws_rht = bf16( sigmoid(x*Wr + ht*Ur + br) * ht ) (bf16)
//   gate 2: ws_xh  = x*Wh + bh                               (f32)
// ---------------------------------------------------------------------------
__global__ __launch_bounds__(NTHREADS) void gru_gates_kernel(
    const unsigned short* __restrict__ xb, const unsigned short* __restrict__ hb,
    const unsigned short* __restrict__ WzT, const unsigned short* __restrict__ UzT,
    const unsigned short* __restrict__ WrT, const unsigned short* __restrict__ UrT,
    const unsigned short* __restrict__ WhT,
    const float* __restrict__ bz, const float* __restrict__ brr, const float* __restrict__ bh,
    const float* __restrict__ ht,
    float* __restrict__ ws_z, float* __restrict__ ws_xh, unsigned short* __restrict__ ws_rht) {
    __shared__ unsigned short sA[2 * BM * LDSS];
    __shared__ unsigned short sB[2 * BN * LDSS];

    const int bc0  = blockIdx.x * BN;
    const int br0  = blockIdx.y * BM;
    const int gate = blockIdx.z;

    v8f acc[4][4];
#pragma unroll
    for (int i = 0; i < 4; ++i)
#pragma unroll
        for (int j = 0; j < 4; ++j)
#pragma unroll
            for (int v = 0; v < 8; ++v) acc[i][j][v] = 0.0f;

    if (gate == 0) {
        gemm_accum(xb, WzT, br0, bc0, sA, sB, acc);
        gemm_accum(hb, UzT, br0, bc0, sA, sB, acc);
    } else if (gate == 1) {
        gemm_accum(xb, WrT, br0, bc0, sA, sB, acc);
        gemm_accum(hb, UrT, br0, bc0, sA, sB, acc);
    } else {
        gemm_accum(xb, WhT, br0, bc0, sA, sB, acc);
    }

    const int lane = threadIdx.x & 31;
    const int w    = threadIdx.x >> 5;
    const int wm = w >> 2, wn = w & 3;
    const int hi = lane >> 4, n0 = lane & 15;

#pragma unroll
    for (int ti = 0; ti < 4; ++ti)
#pragma unroll
        for (int tj = 0; tj < 4; ++tj)
#pragma unroll
            for (int v = 0; v < 8; ++v) {
                const int row = br0 + wm * 64 + ti * 16 + hi * 8 + v;
                const int col = bc0 + wn * 64 + tj * 16 + n0;
                const size_t idx = (size_t)row * HDIM + col;
                const float a = acc[ti][tj][v];
                if (gate == 0) {
                    ws_z[idx] = 1.0f / (1.0f + __expf(-(a + bz[col])));
                } else if (gate == 1) {
                    const float r = 1.0f / (1.0f + __expf(-(a + brr[col])));
                    ws_rht[idx] = f2bf(r * ht[idx]);
                } else {
                    ws_xh[idx] = a + bh[col];
                }
            }
}

// ---------------------------------------------------------------------------
// Phase 2: h~ = tanh((r*ht)*Uh + xh); out = (1-z)*ht + z*h~
// ---------------------------------------------------------------------------
__global__ __launch_bounds__(NTHREADS) void gru_final_kernel(
    const unsigned short* __restrict__ rht, const unsigned short* __restrict__ UhT,
    const float* __restrict__ ws_z, const float* __restrict__ ws_xh,
    const float* __restrict__ ht, float* __restrict__ out) {
    __shared__ unsigned short sA[2 * BM * LDSS];
    __shared__ unsigned short sB[2 * BN * LDSS];

    const int bc0 = blockIdx.x * BN;
    const int br0 = blockIdx.y * BM;

    v8f acc[4][4];
#pragma unroll
    for (int i = 0; i < 4; ++i)
#pragma unroll
        for (int j = 0; j < 4; ++j)
#pragma unroll
            for (int v = 0; v < 8; ++v) acc[i][j][v] = 0.0f;

    gemm_accum(rht, UhT, br0, bc0, sA, sB, acc);

    const int lane = threadIdx.x & 31;
    const int w    = threadIdx.x >> 5;
    const int wm = w >> 2, wn = w & 3;
    const int hi = lane >> 4, n0 = lane & 15;

#pragma unroll
    for (int ti = 0; ti < 4; ++ti)
#pragma unroll
        for (int tj = 0; tj < 4; ++tj)
#pragma unroll
            for (int v = 0; v < 8; ++v) {
                const int row = br0 + wm * 64 + ti * 16 + hi * 8 + v;
                const int col = bc0 + wn * 64 + tj * 16 + n0;
                const size_t idx = (size_t)row * HDIM + col;
                const float hcand = tanhf(acc[ti][tj][v] + ws_xh[idx]);
                const float z = ws_z[idx];
                out[idx] = (1.0f - z) * ht[idx] + z * hcand;
            }
}

// ---------------------------------------------------------------------------
extern "C" void kernel_launch(void* const* d_in, const int* in_sizes, int n_in,
                              void* d_out, int out_size, void* d_ws, size_t ws_size,
                              hipStream_t stream) {
    const float* x   = (const float*)d_in[0];
    const float* ht  = (const float*)d_in[1];
    const float* Wz  = (const float*)d_in[2];
    const float* Uz  = (const float*)d_in[3];
    const float* bz  = (const float*)d_in[4];
    const float* Wr  = (const float*)d_in[5];
    const float* Ur  = (const float*)d_in[6];
    const float* brr = (const float*)d_in[7];
    const float* Wh  = (const float*)d_in[8];
    const float* Uh  = (const float*)d_in[9];
    const float* bh  = (const float*)d_in[10];

    const size_t elems = (size_t)BDIM * HDIM;   // 16.78M
    const size_t welms = (size_t)HDIM * HDIM;   // 4.19M

    char* p = (char*)d_ws;
    float* ws_z  = (float*)p;                 p += elems * 4;
    float* ws_xh = (float*)p;                 p += elems * 4;
    unsigned short* ws_rht = (unsigned short*)p; p += elems * 2;
    unsigned short* xb  = (unsigned short*)p; p += elems * 2;
    unsigned short* hb  = (unsigned short*)p; p += elems * 2;
    unsigned short* WzT = (unsigned short*)p; p += welms * 2;
    unsigned short* UzT = (unsigned short*)p; p += welms * 2;
    unsigned short* WrT = (unsigned short*)p; p += welms * 2;
    unsigned short* UrT = (unsigned short*)p; p += welms * 2;
    unsigned short* WhT = (unsigned short*)p; p += welms * 2;
    unsigned short* UhT = (unsigned short*)p; p += welms * 2;

    dim3 block(NTHREADS);

    // Phase 0: conversions (stream-ordered before GEMMs)
    {
        const int n4 = (int)(elems / 4);
        dim3 cg((n4 + NTHREADS - 1) / NTHREADS);
        convert_bf16_kernel<<<cg, block, 0, stream>>>(x,  xb, n4);
        convert_bf16_kernel<<<cg, block, 0, stream>>>(ht, hb, n4);
        dim3 tg(HDIM / TT, HDIM / TT);
        transpose_bf16_kernel<<<tg, block, 0, stream>>>(Wz, WzT);
        transpose_bf16_kernel<<<tg, block, 0, stream>>>(Uz, UzT);
        transpose_bf16_kernel<<<tg, block, 0, stream>>>(Wr, WrT);
        transpose_bf16_kernel<<<tg, block, 0, stream>>>(Ur, UrT);
        transpose_bf16_kernel<<<tg, block, 0, stream>>>(Wh, WhT);
        transpose_bf16_kernel<<<tg, block, 0, stream>>>(Uh, UhT);
    }

    // Phase 1: gates
    dim3 grid1(HDIM / BN, BDIM / BM, 3);
    gru_gates_kernel<<<grid1, block, 0, stream>>>(xb, hb, WzT, UzT, WrT, UrT, WhT,
                                                  bz, brr, bh, ht,
                                                  ws_z, ws_xh, ws_rht);

    // Phase 2: candidate + blend
    dim3 grid2(HDIM / BN, BDIM / BM, 1);
    gru_final_kernel<<<grid2, block, 0, stream>>>(ws_rht, UhT, ws_z, ws_xh, ht,
                                                  (float*)d_out);
}
